// MyGRU_80994493268527
// MI455X (gfx1250) — matrix operand order repaired and checked by reading, not compiled, not run on hardware
//
#include <hip/hip_runtime.h>
#include <stdint.h>

typedef __attribute__((ext_vector_type(16))) __bf16 v16bf;
typedef __attribute__((ext_vector_type(8)))  float  v8f;

#define H_DIM   128
#define G_DIM   384          // 3*H
#define TILE_M  16           // batch rows per workgroup
#define BDIM    256          // 8 wave32s
#define NT_WAVE 3            // 24 N-tiles / 8 waves

#if defined(__has_builtin)
#  if __has_builtin(__builtin_amdgcn_tensor_load_to_lds)
#    define HAVE_TDM 1
#  else
#    define HAVE_TDM 0
#  endif
#else
#  define HAVE_TDM 0
#endif

__device__ __forceinline__ unsigned short f32_to_bf16(float f) {
    union { float f; uint32_t u; } c; c.f = f;
    uint32_t u = c.u;
    u += 0x7FFFu + ((u >> 16) & 1u);        // round-to-nearest-even
    return (unsigned short)(u >> 16);
}

// Assemble a v16bf A/B fragment from two 16-byte LDS chunks (ds_load_b128 x2)
__device__ __forceinline__ v16bf load_frag(const unsigned short* p0,
                                           const unsigned short* p1) {
    union { uint4 q[2]; v16bf v; } u;
    u.q[0] = *(const uint4*)p0;
    u.q[1] = *(const uint4*)p1;
    return u.v;
}

__device__ __forceinline__ void wait_tensorcnt_0() {
#if __has_builtin(__builtin_amdgcn_s_wait_tensorcnt)
    __builtin_amdgcn_s_wait_tensorcnt(0);
#else
    asm volatile("s_wait_tensorcnt 0x0" ::: "memory");
#endif
}
__device__ __forceinline__ void wait_tensorcnt_1() {
#if __has_builtin(__builtin_amdgcn_s_wait_tensorcnt)
    __builtin_amdgcn_s_wait_tensorcnt(1);
#else
    asm volatile("s_wait_tensorcnt 0x1" ::: "memory");
#endif
}

#if HAVE_TDM
typedef __attribute__((ext_vector_type(4))) unsigned int u32x4;
typedef __attribute__((ext_vector_type(4))) int          i32x4;
typedef __attribute__((ext_vector_type(8))) int          i32x8;

// Issue a TDM load of a (TILE_M x KX) bf16 tile, rows strided T*KX elements,
// into LDS at lds_byte_off. Tracked by TENSORcnt.
template<int KX>
__device__ __forceinline__ void tdm_load_tile(const unsigned short* gsrc,
                                              unsigned int lds_byte_off, int T) {
    unsigned long long ga = (unsigned long long)(uintptr_t)gsrc;
    u32x4 g0;
    g0[0] = 1u;                                                    // count=1 (valid), user D#
    g0[1] = lds_byte_off;                                          // lds_addr
    g0[2] = (unsigned int)ga;                                      // global_addr[31:0]
    g0[3] = (unsigned int)((ga >> 32) & 0x01FFFFFFu) | (2u << 30); // global_addr[56:32] | type=2
    i32x8 g1;
    g1[0] = 1 << 16;            // workgroup_mask=0 | data_size=1 (2 bytes)
    g1[1] = (KX & 0xFFFF) << 16;// tensor_dim0[15:0]  (bits 79:48 -> w1[31:16])
    g1[2] = TILE_M << 16;       // tensor_dim0 hi=0 | tensor_dim1[15:0]=16
    g1[3] = KX << 16;           // tensor_dim1 hi=0 | tile_dim0=KX
    g1[4] = TILE_M;             // tile_dim1=16 | tile_dim2=0
    g1[5] = T * KX;             // tensor_dim0_stride[31:0]
    g1[6] = 0;                  // stride hi | tensor_dim1_stride lo
    g1[7] = 0;
    i32x4 gz = {0, 0, 0, 0};
#if defined(__clang_major__) && (__clang_major__ >= 23)
    i32x8 gz8 = {};
    __builtin_amdgcn_tensor_load_to_lds(g0, g1, gz, gz, gz8, 0);
#else
    __builtin_amdgcn_tensor_load_to_lds(g0, g1, gz, gz, 0);
#endif
}
#endif // HAVE_TDM

template<int KX>
struct __align__(16) LayerSmem {
    unsigned short Wih[G_DIM * KX];      // bf16, row-major (384 x KX)
    unsigned short Whh[G_DIM * H_DIM];   // bf16, row-major (384 x 128)
    unsigned short Xt0[TILE_M * KX];     // double-buffered staged x_t tile (bf16)
    unsigned short Xt1[TILE_M * KX];
    unsigned short Hbf[TILE_M * H_DIM];  // bf16 shadow of h
    float          Hf [TILE_M * H_DIM];  // f32 master h
    float          S1 [TILE_M * G_DIM];  // xr+hr+b | xz+hz+b | xn+bih_n
    float          S2 [TILE_M * H_DIM];  // hn+bhh_n
    float          Bih[G_DIM];
    float          Bhh[G_DIM];
};

// One GRU layer, fused input-projection + recurrent scan; input is bf16.
// Each workgroup owns TILE_M batch rows for all T timesteps. Weights live in
// LDS; the x_t tile for step t+1 streams in via the Tensor Data Mover while
// step t computes.
template<int KX, bool STORE_ALL>
__global__ __launch_bounds__(BDIM)
void gru_layer_kernel(const unsigned short* __restrict__ in,   // (B,T,KX) bf16
                      const float* __restrict__ Wih,           // (384,KX)
                      const float* __restrict__ Whh,           // (384,128)
                      const float* __restrict__ bih,           // (384,)
                      const float* __restrict__ bhh,           // (384,)
                      unsigned short* __restrict__ out_all,    // (B,T,128) bf16
                      float*          __restrict__ out_last,   // (B,128) f32
                      int T)
{
    __shared__ LayerSmem<KX> sm;

    const int tid   = threadIdx.x;
    const int wave  = tid >> 5;
    const int lane  = tid & 31;
    const int lhalf = lane >> 4;      // 0: lanes 0-15, 1: lanes 16-31
    const int lrow  = lane & 15;
    const int row0  = blockIdx.x * TILE_M;

#if HAVE_TDM
    const unsigned int xt_off[2] = { (unsigned int)(uintptr_t)&sm.Xt0[0],
                                     (unsigned int)(uintptr_t)&sm.Xt1[0] };
    // Kick off the t=0 tile immediately (overlaps with weight conversion).
    if (wave == 0)
        tdm_load_tile<KX>(in + (size_t)row0 * T * KX, xt_off[0], T);
#endif

    // ---- one-time: weights -> bf16 LDS, biases, h = 0 ----
    for (int e = tid; e < G_DIM * KX;    e += BDIM) sm.Wih[e] = f32_to_bf16(Wih[e]);
    for (int e = tid; e < G_DIM * H_DIM; e += BDIM) sm.Whh[e] = f32_to_bf16(Whh[e]);
    for (int e = tid; e < G_DIM;         e += BDIM) { sm.Bih[e] = bih[e]; sm.Bhh[e] = bhh[e]; }
    for (int e = tid; e < TILE_M * H_DIM; e += BDIM) { sm.Hf[e] = 0.0f; sm.Hbf[e] = 0; }
    __syncthreads();

    for (int t = 0; t < T; ++t) {
#if HAVE_TDM
        // Wave 0: prefetch tile t+1 into the other buffer, then drain tile t.
        if (wave == 0) {
            if (t + 1 < T) {
                tdm_load_tile<KX>(in + ((size_t)row0 * T + (t + 1)) * KX,
                                  xt_off[(t + 1) & 1], T);
                wait_tensorcnt_1();      // tile t complete; t+1 may stay in flight
            } else {
                wait_tensorcnt_0();
            }
        }
#else
        { // synchronous staging fallback
            unsigned short* xb = (t & 1) ? sm.Xt1 : sm.Xt0;
            #pragma unroll
            for (int e = tid; e < TILE_M * KX; e += BDIM) {
                int m = e / KX, i = e % KX;
                xb[e] = in[((size_t)(row0 + m) * T + t) * KX + i];
            }
        }
#endif
        __syncthreads();
        const unsigned short* xcur = (t & 1) ? sm.Xt1 : sm.Xt0;

        // ---- A-fragments (shared across this wave's 3 N-tiles) ----
        v16bf aH[H_DIM / 32];
        #pragma unroll
        for (int kc = 0; kc < H_DIM / 32; ++kc)
            aH[kc] = load_frag(&sm.Hbf[lrow * H_DIM + kc * 32 + lhalf * 8],
                               &sm.Hbf[lrow * H_DIM + kc * 32 + 16 + lhalf * 8]);
        v16bf aX[KX / 32];
        #pragma unroll
        for (int kc = 0; kc < KX / 32; ++kc)
            aX[kc] = load_frag(&xcur[lrow * KX + kc * 32 + lhalf * 8],
                               &xcur[lrow * KX + kc * 32 + 16 + lhalf * 8]);

        // ---- per-wave N-tiles: gate pre-activations via WMMA ----
        #pragma unroll
        for (int q = 0; q < NT_WAVE; ++q) {
            const int nt = wave * NT_WAVE + q;          // 0..23
            const int ch = nt * 16 + lrow;              // gate channel of this lane
            v8f accX = {}, accH = {};
            #pragma unroll
            for (int kc = 0; kc < KX / 32; ++kc) {
                v16bf b = load_frag(&sm.Wih[(nt * 16 + lrow) * KX + kc * 32 + lhalf * 8],
                                    &sm.Wih[(nt * 16 + lrow) * KX + kc * 32 + 16 + lhalf * 8]);
                accX = __builtin_amdgcn_wmma_f32_16x16x32_bf16(
                           false, aX[kc], false, b, (short)0, accX, false, false);
            }
            #pragma unroll
            for (int kc = 0; kc < H_DIM / 32; ++kc) {
                v16bf b = load_frag(&sm.Whh[(nt * 16 + lrow) * H_DIM + kc * 32 + lhalf * 8],
                                    &sm.Whh[(nt * 16 + lrow) * H_DIM + kc * 32 + 16 + lhalf * 8]);
                accH = __builtin_amdgcn_wmma_f32_16x16x32_bf16(
                           false, aH[kc], false, b, (short)0, accH, false, false);
            }
            // C layout: elem r -> (m = lhalf*8 + r, n = lrow)
            if (ch < 2 * H_DIM) {                       // r,z gates: sum x-side + h-side
                float bsum = sm.Bih[ch] + sm.Bhh[ch];
                #pragma unroll
                for (int r = 0; r < 8; ++r) {
                    int m = lhalf * 8 + r;
                    sm.S1[m * G_DIM + ch] = accX[r] + accH[r] + bsum;
                }
            } else {                                    // n gate: keep sides separate
                float bi = sm.Bih[ch], bh = sm.Bhh[ch];
                #pragma unroll
                for (int r = 0; r < 8; ++r) {
                    int m = lhalf * 8 + r;
                    sm.S1[m * G_DIM + ch]               = accX[r] + bi;
                    sm.S2[m * H_DIM + (ch - 2 * H_DIM)] = accH[r] + bh;
                }
            }
        }
        __syncthreads();

        // ---- gates + state update (f32), 2048 elems / 256 threads ----
        #pragma unroll
        for (int e = tid; e < TILE_M * H_DIM; e += BDIM) {
            int m = e >> 7, j = e & (H_DIM - 1);
            float r  = 1.0f / (1.0f + __expf(-sm.S1[m * G_DIM + j]));
            float z  = 1.0f / (1.0f + __expf(-sm.S1[m * G_DIM + H_DIM + j]));
            float nv = tanhf(sm.S1[m * G_DIM + 2 * H_DIM + j] + r * sm.S2[m * H_DIM + j]);
            float hn = (1.0f - z) * nv + z * sm.Hf[e];
            sm.Hf[e]  = hn;
            sm.Hbf[e] = f32_to_bf16(hn);
            if constexpr (STORE_ALL) {
                out_all[((size_t)(row0 + m) * T + t) * H_DIM + j] = f32_to_bf16(hn);
            } else {
                if (t == T - 1) out_last[(size_t)(row0 + m) * H_DIM + j] = hn;
            }
        }
        __syncthreads();
    }
}

__global__ __launch_bounds__(256)
void cast_f32_to_bf16_kernel(const float* __restrict__ src,
                             unsigned short* __restrict__ dst, size_t n)
{
    size_t i = (size_t)blockIdx.x * blockDim.x + threadIdx.x;
    size_t stride = (size_t)gridDim.x * blockDim.x;
    for (; i < n; i += stride) dst[i] = f32_to_bf16(src[i]);
}

__global__ __launch_bounds__(256)
void out_proj_kernel(const float* __restrict__ h,     // (B,128)
                     const float* __restrict__ Wout,  // (1,128)
                     const float* __restrict__ bout,  // (1,)
                     float* __restrict__ out, int B)
{
    int b = blockIdx.x * blockDim.x + threadIdx.x;
    if (b < B) {
        float acc = 0.0f;
        #pragma unroll
        for (int j = 0; j < H_DIM; ++j) acc += h[(size_t)b * H_DIM + j] * Wout[j];
        out[b] = acc + bout[0];
    }
}

extern "C" void kernel_launch(void* const* d_in, const int* in_sizes, int n_in,
                              void* d_out, int out_size, void* d_ws, size_t ws_size,
                              hipStream_t stream)
{
    const float* x    = (const float*)d_in[0];
    const float* Wih0 = (const float*)d_in[1];
    const float* Whh0 = (const float*)d_in[2];
    const float* bih0 = (const float*)d_in[3];
    const float* bhh0 = (const float*)d_in[4];
    const float* Wih1 = (const float*)d_in[5];
    const float* Whh1 = (const float*)d_in[6];
    const float* bih1 = (const float*)d_in[7];
    const float* bhh1 = (const float*)d_in[8];
    const float* Wout = (const float*)d_in[9];
    const float* bout = (const float*)d_in[10];

    const int I = 64, T = 1024;
    const int B = in_sizes[0] / (T * I);               // 256

    // workspace layout: x_bf16 (B,T,64) | h1 (B,T,128) bf16 | h2_last (B,128) f32
    unsigned short* xbf = (unsigned short*)d_ws;
    unsigned short* h1  = xbf + (size_t)B * T * I;
    float* h2last = (float*)(h1 + (size_t)B * T * H_DIM);

    cast_f32_to_bf16_kernel<<<dim3(2048), dim3(256), 0, stream>>>(
        x, xbf, (size_t)B * T * I);

    dim3 grid(B / TILE_M), block(BDIM);

    gru_layer_kernel<64, true><<<grid, block, 0, stream>>>(
        xbf, Wih0, Whh0, bih0, bhh0, h1, (float*)nullptr, T);

    gru_layer_kernel<128, false><<<grid, block, 0, stream>>>(
        h1, Wih1, Whh1, bih1, bhh1, (unsigned short*)nullptr, h2last, T);

    out_proj_kernel<<<dim3((B + 255) / 256), dim3(256), 0, stream>>>(
        h2last, Wout, bout, (float*)d_out, B);
}
